// PointsFusion_64201171140764
// MI455X (gfx1250) — matrix operand order repaired and checked by reading, not compiled, not run on hardware
//
#include <hip/hip_runtime.h>

// ---------------------------------------------------------------------------
// Problem constants (t = 0.5 -> static split sizes per the reference)
// ---------------------------------------------------------------------------
#define BATCH   8
#define NPTS    8192
#define NOUT    2048
#define NHALF   1024     // N1 = N2 = 1024
#define KNN     32
#define KHALF   16       // k1 = k2 = 16
#define C2      64
#define C3      128
#define MROWS   (BATCH * NOUT * KNN)     // 524288
#define MTILES  (MROWS / 16)             // 32768
#define BN_EPS  1e-3f

// Workspace layout (bytes, 256-aligned)
#define OFF_NEWPTS  0u
#define OFF_FEATS   262144u
#define OFF_GRP     8650752u
#define OFF_SCORES  14942208u
#define OFF_W1      17039360u
#define OFF_B1      17040384u
#define OFF_B2      17040640u
#define OFF_B3      17040896u
#define OFF_W2H     17041408u
#define OFF_W3H     17049600u

typedef __attribute__((ext_vector_type(16))) _Float16 v16h;
typedef __attribute__((ext_vector_type(8)))  float    v8f;

static __device__ inline v16h load16h(const _Float16* p) {
    v16h r;
#pragma unroll
    for (int i = 0; i < 16; ++i) r[i] = p[i];
    return r;
}

// ---------------------------------------------------------------------------
// Kernel 1: fold BN into conv weights/bias; pre-swizzle W2/W3 into the
// CDNA5 WMMA 16-bit B-matrix (32x16, K-major packed pairs) lane layout.
// ---------------------------------------------------------------------------
__global__ __launch_bounds__(256) void prep_kernel(
    const float* w1, const float* b1, const float* g1, const float* be1, const float* mu1, const float* va1,
    const float* w2, const float* b2, const float* g2, const float* be2, const float* mu2, const float* va2,
    const float* w3, const float* b3, const float* g3, const float* be3, const float* mu3, const float* va3,
    float* W1f, float* B1f, float* B2f, float* B3f, _Float16* W2H, _Float16* W3H)
{
    const int tid = threadIdx.x;
    if (tid < 64) {
        float s1 = g1[tid] * rsqrtf(va1[tid] + BN_EPS);
#pragma unroll
        for (int c = 0; c < 4; ++c) W1f[tid * 4 + c] = w1[tid * 4 + c] * s1;
        B1f[tid] = (b1[tid] - mu1[tid]) * s1 + be1[tid];
        float s2 = g2[tid] * rsqrtf(va2[tid] + BN_EPS);
        B2f[tid] = (b2[tid] - mu2[tid]) * s2 + be2[tid];
    }
    if (tid < 128) {
        float s3 = g3[tid] * rsqrtf(va3[tid] + BN_EPS);
        B3f[tid] = (b3[tid] - mu3[tid]) * s3 + be3[tid];
    }
    // W2 swizzle: 8 blocks (nt 0..3, kb 0..1) x 32 lanes x 16 halfs
    {
        int blk = tid >> 5, lane = tid & 31;
        int nt = blk >> 1, kb = blk & 1;
        int n = nt * 16 + (lane & 15);
        float s = g2[n] * rsqrtf(va2[n] + BN_EPS);
#pragma unroll
        for (int j = 0; j < 16; ++j) {
            int k = kb * 32 + (lane >> 4) * 8 + (j & 7) + (j >> 3) * 16;
            W2H[blk * 512 + lane * 16 + j] = (_Float16)(w2[n * 64 + k] * s);
        }
    }
    // W3 swizzle: 16 blocks (nt 0..7, kb 0..1)
#pragma unroll
    for (int it = 0; it < 2; ++it) {
        int blk = it * 8 + (tid >> 5), lane = tid & 31;
        int nt = blk >> 1, kb = blk & 1;
        int n = nt * 16 + (lane & 15);
        float s = g3[n] * rsqrtf(va3[n] + BN_EPS);
#pragma unroll
        for (int j = 0; j < 16; ++j) {
            int k = kb * 32 + (lane >> 4) * 8 + (j & 7) + (j >> 3) * 16;
            W3H[blk * 512 + lane * 16 + j] = (_Float16)(w3[n * 64 + k] * s);
        }
    }
}

// ---------------------------------------------------------------------------
// Kernel 2: furthest point sampling. One 1024-thread block per (batch,cloud).
// Outputs sampled points as float4 {x,y,z,|p|^2}.
// ---------------------------------------------------------------------------
__global__ __launch_bounds__(1024) void fps_kernel(const float* p1, const float* p2,
                                                   float* newpts)
{
    const int blk = blockIdx.x;
    const int batch = blk >> 1, cloud = blk & 1;
    const float* src = cloud ? p2 : p1;
    const float* bx = src + (size_t)batch * 3 * NPTS;
    const float* by = bx + NPTS;
    const float* bz = bx + 2 * NPTS;
    const int tid = threadIdx.x;

    float px[8], py[8], pz[8], dd[8];
#pragma unroll
    for (int u = 0; u < 8; ++u) {
        int i = u * 1024 + tid;
        px[u] = bx[i]; py[u] = by[i]; pz[u] = bz[i];
        dd[u] = __builtin_inff();
    }

    __shared__ float scx, scy, scz;
    __shared__ float rv[32];
    __shared__ int   ri[32];
    __shared__ int   sIdx;

    float4* out = (float4*)newpts;
    const int qbase = batch * NOUT + cloud * NHALF;

    if (tid == 0) {
        scx = px[0]; scy = py[0]; scz = pz[0];
        out[qbase] = make_float4(px[0], py[0], pz[0],
                                 px[0]*px[0] + py[0]*py[0] + pz[0]*pz[0]);
    }
    __syncthreads();

    for (int s = 1; s < NHALF; ++s) {
        float cx = scx, cy = scy, cz = scz;
        float best = -__builtin_inff();
        int bidx = 0x7fffffff;
#pragma unroll
        for (int u = 0; u < 8; ++u) {
            float dx = px[u] - cx, dy = py[u] - cy, dz = pz[u] - cz;
            float d = dx*dx + dy*dy + dz*dz;
            dd[u] = fminf(dd[u], d);
            if (dd[u] > best) { best = dd[u]; bidx = u * 1024 + tid; }
        }
#pragma unroll
        for (int mask = 1; mask < 32; mask <<= 1) {
            float ov = __shfl_xor(best, mask);
            int   oi = __shfl_xor(bidx, mask);
            if (ov > best || (ov == best && oi < bidx)) { best = ov; bidx = oi; }
        }
        int wid = tid >> 5;
        if ((tid & 31) == 0) { rv[wid] = best; ri[wid] = bidx; }
        __syncthreads();
        if (tid < 32) {
            best = rv[tid]; bidx = ri[tid];
#pragma unroll
            for (int mask = 1; mask < 32; mask <<= 1) {
                float ov = __shfl_xor(best, mask);
                int   oi = __shfl_xor(bidx, mask);
                if (ov > best || (ov == best && oi < bidx)) { best = ov; bidx = oi; }
            }
            if (tid == 0) sIdx = bidx;
        }
        __syncthreads();
        int idx = sIdx;
        if ((idx & 1023) == tid) {
            int u = idx >> 10;
            float sx = 0.f, sy = 0.f, sz = 0.f;
#pragma unroll
            for (int u2 = 0; u2 < 8; ++u2)
                if (u2 == u) { sx = px[u2]; sy = py[u2]; sz = pz[u2]; }
            scx = sx; scy = sy; scz = sz;
            out[qbase + s] = make_float4(sx, sy, sz, sx*sx + sy*sy + sz*sz);
        }
        __syncthreads();
    }
}

// ---------------------------------------------------------------------------
// Kernel 3: kNN (top-16 per half, stable tie-break on index) + features.
// feats[row] = {resi.xyz, ||resi||}; grp[row] = neighbor xyz.
// One thread per query; 2048 refs cached in LDS.
// ---------------------------------------------------------------------------
__global__ __launch_bounds__(256) void knn_kernel(const float* newpts,
                                                  float* feats, float* grp)
{
    __shared__ float4 sref[NOUT];
    const int batch = blockIdx.x >> 3;
    const int n = ((blockIdx.x & 7) << 8) + threadIdx.x;
    const float4* np4 = (const float4*)newpts + (size_t)batch * NOUT;
    for (int i = threadIdx.x; i < NOUT; i += 256) sref[i] = np4[i];
    __syncthreads();

    const float4 q = sref[n];
    const size_t rowbase = ((size_t)batch * NOUT + n) * KNN;
    float4* f4 = (float4*)feats;

    for (int half = 0; half < 2; ++half) {
        float bd[16];
        int   bi[16];
#pragma unroll
        for (int j = 0; j < 16; ++j) { bd[j] = __builtin_inff(); bi[j] = 0x7fffffff; }
        const int rb = half << 10;
        for (int r = 0; r < NHALF; ++r) {
            float4 p = sref[rb + r];
            float d2 = q.w + p.w - 2.0f * (q.x*p.x + q.y*p.y + q.z*p.z);
            if (d2 < bd[15]) {
                float v = d2; int ix = r;
#pragma unroll
                for (int sl = 0; sl < 16; ++sl) {
                    bool better = (v < bd[sl]) || (v == bd[sl] && ix < bi[sl]);
                    float tv = bd[sl]; int ti = bi[sl];
                    if (better) { bd[sl] = v; bi[sl] = ix; v = tv; ix = ti; }
                }
            }
        }
#pragma unroll
        for (int j = 0; j < 16; ++j) {
            float4 p = sref[rb + bi[j]];
            float rx = p.x - q.x, ry = p.y - q.y, rz = p.z - q.z;
            float d2s = rx*rx + ry*ry + rz*rz;
            float dist = (d2s > 0.f) ? sqrtf(d2s) : 0.f;
            size_t row = rowbase + half * KHALF + j;
            f4[row] = make_float4(rx, ry, rz, dist);
            grp[row * 3 + 0] = p.x;
            grp[row * 3 + 1] = p.y;
            grp[row * 3 + 2] = p.z;
        }
    }
}

// ---------------------------------------------------------------------------
// Kernel 4: fused MLP (L1 VALU, L2+L3 via v_wmma_f32_16x16x32_f16) + channel
// max. One wave per 16-row M-tile; L2 activations bounce through a 2KB/wave
// LDS region to convert the D-tile layout into A-fragment layout for L3.
// ---------------------------------------------------------------------------
__global__ __launch_bounds__(256) void mlp_kernel(const float* feats,
                                                  const float* W1f, const float* B1f,
                                                  const float* B2f, const float* B3f,
                                                  const _Float16* W2H, const _Float16* W3H,
                                                  float* scores)
{
    __shared__ float    sW1[256];
    __shared__ float    sB1[64], sB2[64], sB3[128];
    __shared__ _Float16 sW2[8 * 512];
    __shared__ _Float16 sW3[16 * 512];
    __shared__ _Float16 sA2[8][1024];

    const int tid = threadIdx.x;
    for (int i = tid; i < 256; i += 256)  sW1[i] = W1f[i];
    if (tid < 64)  { sB1[tid] = B1f[tid]; sB2[tid] = B2f[tid]; }
    if (tid < 128) { sB3[tid] = B3f[tid]; }
    for (int i = tid; i < 4096; i += 256) sW2[i] = W2H[i];
    for (int i = tid; i < 8192; i += 256) sW3[i] = W3H[i];
    __syncthreads();

    const int wv = tid >> 5;
    const int l  = tid & 31;
    const int r  = l & 15;
    const int h  = l >> 4;
    const int t  = blockIdx.x * 8 + wv;

    // ---- Layer 1 (K=4): compute A-fragments directly in register layout ----
    const float4 f = ((const float4*)feats)[t * 16 + r];
    v16h a0, a1;
#pragma unroll
    for (int fg = 0; fg < 2; ++fg) {
#pragma unroll
        for (int i = 0; i < 8; ++i) {
#pragma unroll
            for (int s = 0; s < 2; ++s) {
                int k = fg * 32 + (i >= 4 ? 16 : 0) + ((i & 3) * 2 + s) + h * 8;
                float v = sW1[k*4+0]*f.x + sW1[k*4+1]*f.y + sW1[k*4+2]*f.z +
                          sW1[k*4+3]*f.w + sB1[k];
                v = fmaxf(v, 0.f);
                if (fg == 0) a0[i * 2 + s] = (_Float16)v;
                else         a1[i * 2 + s] = (_Float16)v;
            }
        }
    }

    // ---- Layer 2: 4 N-tiles x (2 WMMA, K=64); scatter to LDS in A-swizzle ----
#pragma unroll
    for (int nt = 0; nt < 4; ++nt) {
        v8f acc = {};
        v16h bf0 = load16h(&sW2[(nt * 2 + 0) * 512 + l * 16]);
        v16h bf1 = load16h(&sW2[(nt * 2 + 1) * 512 + l * 16]);
        acc = __builtin_amdgcn_wmma_f32_16x16x32_f16(false, a0, false, bf0,
                                                     (short)0, acc, false, false);
        acc = __builtin_amdgcn_wmma_f32_16x16x32_f16(false, a1, false, bf1,
                                                     (short)0, acc, false, false);
        const int nc = nt * 16 + r;          // output channel (k of layer 3)
        const float bb = sB2[nc];
        const int kk = nc & 31, fgi = nc >> 5;
        const int lp  = (((kk >> 3) & 1) << 4);
        const int pos = (((kk & 16) >> 2) + ((kk & 7) >> 1)) * 2 + (kk & 1);
#pragma unroll
        for (int v = 0; v < 8; ++v) {
            int ml = v + 8 * h;              // row within tile
            float val = fmaxf(acc[v] + bb, 0.f);
            sA2[wv][fgi * 512 + (ml + lp) * 16 + pos] = (_Float16)val;
        }
    }
    __syncthreads();

    v16h c0 = load16h(&sA2[wv][0   + l * 16]);
    v16h c1 = load16h(&sA2[wv][512 + l * 16]);

    // ---- Layer 3: 8 N-tiles x (2 WMMA); running max over channels ----
    float mx[8];
#pragma unroll
    for (int v = 0; v < 8; ++v) mx[v] = 0.f;   // ReLU outputs are >= 0
#pragma unroll
    for (int nt = 0; nt < 8; ++nt) {
        v8f acc = {};
        v16h bf0 = load16h(&sW3[(nt * 2 + 0) * 512 + l * 16]);
        v16h bf1 = load16h(&sW3[(nt * 2 + 1) * 512 + l * 16]);
        acc = __builtin_amdgcn_wmma_f32_16x16x32_f16(false, c0, false, bf0,
                                                     (short)0, acc, false, false);
        acc = __builtin_amdgcn_wmma_f32_16x16x32_f16(false, c1, false, bf1,
                                                     (short)0, acc, false, false);
        const float bb = sB3[nt * 16 + r];
#pragma unroll
        for (int v = 0; v < 8; ++v) mx[v] = fmaxf(mx[v], acc[v] + bb);
    }
    // max over the 16 columns held across each 16-lane half
#pragma unroll
    for (int mask = 1; mask < 16; mask <<= 1)
#pragma unroll
        for (int v = 0; v < 8; ++v)
            mx[v] = fmaxf(mx[v], __shfl_xor(mx[v], mask));
    if (r == 0) {
#pragma unroll
        for (int v = 0; v < 8; ++v)
            scores[(size_t)t * 16 + v + 8 * h] = mx[v];
    }
}

// ---------------------------------------------------------------------------
// Kernel 5: softmax over k=32 neighbors + weighted fusion. One wave / point.
// out layout [B,3,NOUT].
// ---------------------------------------------------------------------------
__global__ __launch_bounds__(256) void fuse_kernel(const float* scores,
                                                   const float* grp, float* out)
{
    const int p = blockIdx.x * 8 + (threadIdx.x >> 5);
    const int l = threadIdx.x & 31;
    const int b = p >> 11, n = p & 2047;
    const size_t row = (size_t)p * KNN + l;

    float s  = scores[row];
    float g0 = grp[row * 3 + 0];
    float g1 = grp[row * 3 + 1];
    float g2 = grp[row * 3 + 2];

    float m = s;
#pragma unroll
    for (int mask = 1; mask < 32; mask <<= 1) m = fmaxf(m, __shfl_xor(m, mask));
    float e = __expf(s - m);
    float sum = e;
#pragma unroll
    for (int mask = 1; mask < 32; mask <<= 1) sum += __shfl_xor(sum, mask);
    const float w = e / sum;

    float a0 = w * g0, a1 = w * g1, a2 = w * g2;
#pragma unroll
    for (int mask = 1; mask < 32; mask <<= 1) {
        a0 += __shfl_xor(a0, mask);
        a1 += __shfl_xor(a1, mask);
        a2 += __shfl_xor(a2, mask);
    }
    if (l == 0) {
        out[((size_t)b * 3 + 0) * NOUT + n] = a0;
        out[((size_t)b * 3 + 1) * NOUT + n] = a1;
        out[((size_t)b * 3 + 2) * NOUT + n] = a2;
    }
}

// ---------------------------------------------------------------------------
extern "C" void kernel_launch(void* const* d_in, const int* in_sizes, int n_in,
                              void* d_out, int out_size, void* d_ws, size_t ws_size,
                              hipStream_t stream)
{
    (void)in_sizes; (void)n_in; (void)out_size; (void)ws_size;
    const float* points1 = (const float*)d_in[0];
    const float* points2 = (const float*)d_in[1];
    // d_in[2] = t (static 0.5 -> hardcoded splits)
    const float* w1  = (const float*)d_in[3];
    const float* b1  = (const float*)d_in[4];
    const float* g1  = (const float*)d_in[5];
    const float* be1 = (const float*)d_in[6];
    const float* mu1 = (const float*)d_in[7];
    const float* va1 = (const float*)d_in[8];
    const float* w2  = (const float*)d_in[9];
    const float* b2  = (const float*)d_in[10];
    const float* g2  = (const float*)d_in[11];
    const float* be2 = (const float*)d_in[12];
    const float* mu2 = (const float*)d_in[13];
    const float* va2 = (const float*)d_in[14];
    const float* w3  = (const float*)d_in[15];
    const float* b3  = (const float*)d_in[16];
    const float* g3  = (const float*)d_in[17];
    const float* be3 = (const float*)d_in[18];
    const float* mu3 = (const float*)d_in[19];
    const float* va3 = (const float*)d_in[20];

    char* ws = (char*)d_ws;
    float*     newpts = (float*)(ws + OFF_NEWPTS);
    float*     feats  = (float*)(ws + OFF_FEATS);
    float*     grp    = (float*)(ws + OFF_GRP);
    float*     sco    = (float*)(ws + OFF_SCORES);
    float*     W1f    = (float*)(ws + OFF_W1);
    float*     B1f    = (float*)(ws + OFF_B1);
    float*     B2f    = (float*)(ws + OFF_B2);
    float*     B3f    = (float*)(ws + OFF_B3);
    _Float16*  W2H    = (_Float16*)(ws + OFF_W2H);
    _Float16*  W3H    = (_Float16*)(ws + OFF_W3H);

    prep_kernel<<<1, 256, 0, stream>>>(w1, b1, g1, be1, mu1, va1,
                                       w2, b2, g2, be2, mu2, va2,
                                       w3, b3, g3, be3, mu3, va3,
                                       W1f, B1f, B2f, B3f, W2H, W3H);
    fps_kernel<<<BATCH * 2, 1024, 0, stream>>>(points1, points2, newpts);
    knn_kernel<<<BATCH * (NOUT / 256), 256, 0, stream>>>(newpts, feats, grp);
    mlp_kernel<<<MTILES / 8, 256, 0, stream>>>(feats, W1f, B1f, B2f, B3f,
                                               W2H, W3H, sco);
    fuse_kernel<<<(BATCH * NOUT) / 8, 256, 0, stream>>>(sco, grp, (float*)d_out);
}
